// DSAttentionLayer_32573031973597
// MI455X (gfx1250) — matrix operand order repaired and checked by reading, not compiled, not run on hardware
//
#include <hip/hip_runtime.h>

typedef __attribute__((ext_vector_type(16))) __bf16 v16bf;
typedef __attribute__((ext_vector_type(8)))  __bf16 v8bf;
typedef __attribute__((ext_vector_type(8)))  float  v8f;

#define D_MODEL 512
#define NHEADS  8
#define HDIM    64
#define BATCH   4
#define SEQ     2048

__device__ __forceinline__ v16bf cat16(v8bf lo, v8bf hi) {
    return __builtin_shufflevector(lo, hi, 0,1,2,3,4,5,6,7,8,9,10,11,12,13,14,15);
}

__device__ __forceinline__ v8f wmma_bf16(v16bf a, v16bf b, v8f c) {
    return __builtin_amdgcn_wmma_f32_16x16x32_bf16(false, a, false, b, (short)0, c, false, false);
}

// C = A[M,512] @ W[512,512] + bias, tiled 64x64 per block, 4 waves of 32x32.
// A is fp32 (projections) or bf16 (final projection of attention context).
// Output bf16 (workspace) or fp32 (d_out).
template<bool A_BF16, bool OUT_F32>
__global__ __launch_bounds__(128)
void gemm512_kernel(const float* __restrict__ A32, const __bf16* __restrict__ A16,
                    const float* __restrict__ W, const float* __restrict__ bias,
                    __bf16* __restrict__ outB, float* __restrict__ outF)
{
    __shared__ __bf16 sA[64][40];   // [m][k], 16B-aligned rows
    __shared__ __bf16 sW[64][40];   // transposed weight tile: [n][k]

    const int t    = threadIdx.x;
    const int lane = t & 31;
    const int wv   = t >> 5;
    const int half = lane >> 4;
    const int l16  = lane & 15;

    const int n0 = blockIdx.x * 64;
    const int m0 = blockIdx.y * 64;
    const int wm = (wv >> 1) * 32;
    const int wn = (wv & 1) * 32;

    v8f acc[2][2];
    #pragma unroll
    for (int i = 0; i < 2; i++)
        #pragma unroll
        for (int j = 0; j < 2; j++)
            #pragma unroll
            for (int r = 0; r < 8; r++) acc[i][j][r] = 0.f;

    for (int k0 = 0; k0 < D_MODEL; k0 += 32) {
        __syncthreads();
        // stage A tile [64 x 32] -> bf16 LDS
        {
            const int cg    = (t & 7) * 4;
            const int rbase = t >> 3;           // 0..15
            #pragma unroll
            for (int i = 0; i < 4; i++) {
                const int r = rbase + i * 16;
                if constexpr (A_BF16) {
                    const __bf16* p = A16 + (size_t)(m0 + r) * D_MODEL + k0 + cg;
                    sA[r][cg+0] = p[0]; sA[r][cg+1] = p[1];
                    sA[r][cg+2] = p[2]; sA[r][cg+3] = p[3];
                } else {
                    const float4 v = *(const float4*)(A32 + (size_t)(m0 + r) * D_MODEL + k0 + cg);
                    sA[r][cg+0] = (__bf16)v.x; sA[r][cg+1] = (__bf16)v.y;
                    sA[r][cg+2] = (__bf16)v.z; sA[r][cg+3] = (__bf16)v.w;
                }
            }
        }
        // stage W tile [32 x 64] transposed -> sW[n][k]
        {
            const int ng = (t & 15) * 4;
            const int kb = t >> 4;              // 0..7
            #pragma unroll
            for (int i = 0; i < 4; i++) {
                const int k = kb + i * 8;
                const float4 v = *(const float4*)(W + (size_t)(k0 + k) * D_MODEL + n0 + ng);
                sW[ng+0][k] = (__bf16)v.x; sW[ng+1][k] = (__bf16)v.y;
                sW[ng+2][k] = (__bf16)v.z; sW[ng+3][k] = (__bf16)v.w;
            }
        }
        __syncthreads();

        v16bf af[2], bfm[2];
        #pragma unroll
        for (int mi = 0; mi < 2; mi++) {        // A layout: lane=M, halves split K
            const int m = wm + mi * 16 + l16;
            v8bf lo = *(const v8bf*)&sA[m][half * 8];
            v8bf hi = *(const v8bf*)&sA[m][16 + half * 8];
            af[mi] = cat16(lo, hi);
        }
        #pragma unroll
        for (int ni = 0; ni < 2; ni++) {        // B layout: lane=N, halves split K
            const int n = wn + ni * 16 + l16;
            v8bf lo = *(const v8bf*)&sW[n][half * 8];
            v8bf hi = *(const v8bf*)&sW[n][16 + half * 8];
            bfm[ni] = cat16(lo, hi);
        }
        #pragma unroll
        for (int mi = 0; mi < 2; mi++)
            #pragma unroll
            for (int ni = 0; ni < 2; ni++)
                acc[mi][ni] = wmma_bf16(af[mi], bfm[ni], acc[mi][ni]);
    }

    // epilogue: C layout lane=N, VGPR r => row r + 8*half
    #pragma unroll
    for (int mi = 0; mi < 2; mi++) {
        #pragma unroll
        for (int ni = 0; ni < 2; ni++) {
            const int col = n0 + wn + ni * 16 + l16;
            const float bvs = bias[col];
            #pragma unroll
            for (int r = 0; r < 8; r++) {
                const int row = m0 + wm + mi * 16 + half * 8 + r;
                const float v = acc[mi][ni][r] + bvs;
                if constexpr (OUT_F32) outF[(size_t)row * D_MODEL + col] = v;
                else                   outB[(size_t)row * D_MODEL + col] = (__bf16)v;
            }
        }
    }
}

// Flash attention: block = (b, h, 64 q-rows); 4 waves x 16 rows each.
__global__ __launch_bounds__(128)
void attn_kernel(const __bf16* __restrict__ Q, const __bf16* __restrict__ K,
                 const __bf16* __restrict__ V, const float* __restrict__ tau,
                 const float* __restrict__ delta, __bf16* __restrict__ Ctx)
{
    __shared__ __bf16 sVt[64][72];      // V tile transposed: [e][s]
    __shared__ __bf16 sP[4][16][40];    // per-wave P tile: [m][s]

    const int t    = threadIdx.x;
    const int lane = t & 31;
    const int wv   = t >> 5;
    const int half = lane >> 4;
    const int l16  = lane & 15;

    const int b  = blockIdx.z;
    const int h  = blockIdx.y;
    const int q0 = blockIdx.x * 64;

    const __bf16* Qb = Q + (size_t)b * SEQ * D_MODEL + (size_t)h * HDIM;
    const __bf16* Kb = K + (size_t)b * SEQ * D_MODEL + (size_t)h * HDIM;
    const __bf16* Vb = V + (size_t)b * SEQ * D_MODEL + (size_t)h * HDIM;

    const float ts = tau[b]   * 0.125f;   // scale = 1/sqrt(64)
    const float dl = delta[b] * 0.125f;

    // preload Q fragments (A layout), E=64 -> two K=32 chunks
    v16bf qf[2];
    {
        const int m = q0 + wv * 16 + l16;
        const __bf16* qp = Qb + (size_t)m * D_MODEL;
        #pragma unroll
        for (int ec = 0; ec < 2; ec++) {
            v8bf lo = *(const v8bf*)(qp + ec * 32 + half * 8);
            v8bf hi = *(const v8bf*)(qp + ec * 32 + 16 + half * 8);
            qf[ec] = cat16(lo, hi);
        }
    }

    v8f oacc[4];
    #pragma unroll
    for (int et = 0; et < 4; et++)
        #pragma unroll
        for (int r = 0; r < 8; r++) oacc[et][r] = 0.f;
    float mrun[8], lrun[8];
    #pragma unroll
    for (int r = 0; r < 8; r++) { mrun[r] = -1e30f; lrun[r] = 0.f; }

    for (int s0 = 0; s0 < SEQ; s0 += 64) {
        __syncthreads();
        // stage V tile [64 x 64] transposed into LDS
        {
            const int eg = (t & 15) * 4;
            const int sb = t >> 4;          // 0..7
            #pragma unroll
            for (int i = 0; i < 8; i++) {
                const int s = sb + i * 8;
                const __bf16* p = Vb + (size_t)(s0 + s) * D_MODEL + eg;
                sVt[eg+0][s] = p[0]; sVt[eg+1][s] = p[1];
                sVt[eg+2][s] = p[2]; sVt[eg+3][s] = p[3];
            }
        }
        __syncthreads();

        // scores S[16 x 64] = Q K^T (K fragments straight from global, B layout)
        v8f sacc[4];
        #pragma unroll
        for (int sub = 0; sub < 4; sub++)
            #pragma unroll
            for (int r = 0; r < 8; r++) sacc[sub][r] = 0.f;
        #pragma unroll
        for (int sub = 0; sub < 4; sub++) {
            const int srow = s0 + sub * 16 + l16;
            const __bf16* kp = Kb + (size_t)srow * D_MODEL;
            #pragma unroll
            for (int ec = 0; ec < 2; ec++) {
                v8bf lo = *(const v8bf*)(kp + ec * 32 + half * 8);
                v8bf hi = *(const v8bf*)(kp + ec * 32 + 16 + half * 8);
                sacc[sub] = wmma_bf16(qf[ec], cat16(lo, hi), sacc[sub]);
            }
        }

        // affine + online softmax (row = half*8 + r; columns across 16-lane half)
        #pragma unroll
        for (int sub = 0; sub < 4; sub++)
            #pragma unroll
            for (int r = 0; r < 8; r++)
                sacc[sub][r] = sacc[sub][r] * ts + dl;

        float psum[8];
        #pragma unroll
        for (int r = 0; r < 8; r++) {
            float mx = fmaxf(fmaxf(sacc[0][r], sacc[1][r]), fmaxf(sacc[2][r], sacc[3][r]));
            mx = fmaxf(mx, __shfl_xor(mx, 1, 16));
            mx = fmaxf(mx, __shfl_xor(mx, 2, 16));
            mx = fmaxf(mx, __shfl_xor(mx, 4, 16));
            mx = fmaxf(mx, __shfl_xor(mx, 8, 16));
            const float mn    = fmaxf(mrun[r], mx);
            const float alpha = __expf(mrun[r] - mn);
            mrun[r] = mn;
            lrun[r] *= alpha;
            #pragma unroll
            for (int et = 0; et < 4; et++) oacc[et][r] *= alpha;
            psum[r] = 0.f;
        }

        #pragma unroll
        for (int sub = 0; sub < 4; sub++) {
            #pragma unroll
            for (int r = 0; r < 8; r++) {
                const float p = __expf(sacc[sub][r] - mrun[r]);
                psum[r] += p;
                sP[wv][half * 8 + r][sub * 16 + l16] = (__bf16)p;  // C -> row-major LDS
            }
        }
        #pragma unroll
        for (int r = 0; r < 8; r++) {
            float tot = psum[r];
            tot += __shfl_xor(tot, 1, 16);
            tot += __shfl_xor(tot, 2, 16);
            tot += __shfl_xor(tot, 4, 16);
            tot += __shfl_xor(tot, 8, 16);
            lrun[r] += tot;
        }

        // O += P @ V  (P re-read from LDS as A layout; same-wave LDS ops are in order)
        #pragma unroll
        for (int sc = 0; sc < 2; sc++) {
            v8bf plo = *(const v8bf*)&sP[wv][l16][sc * 32 + half * 8];
            v8bf phi = *(const v8bf*)&sP[wv][l16][sc * 32 + 16 + half * 8];
            const v16bf pf = cat16(plo, phi);
            #pragma unroll
            for (int et = 0; et < 4; et++) {
                v8bf vlo = *(const v8bf*)&sVt[et * 16 + l16][sc * 32 + half * 8];
                v8bf vhi = *(const v8bf*)&sVt[et * 16 + l16][sc * 32 + 16 + half * 8];
                oacc[et] = wmma_bf16(pf, cat16(vlo, vhi), oacc[et]);
            }
        }
    }

    // normalize and store context (bf16) in [B*L, D] layout, head columns h*64..
    #pragma unroll
    for (int r = 0; r < 8; r++) {
        const float inv = 1.0f / lrun[r];
        const int row = b * SEQ + q0 + wv * 16 + half * 8 + r;
        #pragma unroll
        for (int et = 0; et < 4; et++) {
            const int col = h * HDIM + et * 16 + l16;
            Ctx[(size_t)row * D_MODEL + col] = (__bf16)(oacc[et][r] * inv);
        }
    }
}

extern "C" void kernel_launch(void* const* d_in, const int* in_sizes, int n_in,
                              void* d_out, int out_size, void* d_ws, size_t ws_size,
                              hipStream_t stream)
{
    const float* queries = (const float*)d_in[0];
    const float* keys    = (const float*)d_in[1];
    const float* values  = (const float*)d_in[2];
    const float* tau     = (const float*)d_in[3];
    const float* delta   = (const float*)d_in[4];
    const float* Wq = (const float*)d_in[5];
    const float* bq = (const float*)d_in[6];
    const float* Wk = (const float*)d_in[7];
    const float* bk = (const float*)d_in[8];
    const float* Wv = (const float*)d_in[9];
    const float* bv = (const float*)d_in[10];
    const float* Wo = (const float*)d_in[11];
    const float* bo = (const float*)d_in[12];
    float* out = (float*)d_out;

    const int M = BATCH * SEQ;                     // 8192
    const size_t plane = (size_t)M * D_MODEL;      // 4M elems (bf16 -> 8MB)
    __bf16* Qbf = (__bf16*)d_ws;
    __bf16* Kbf = Qbf + plane;
    __bf16* Vbf = Kbf + plane;
    __bf16* Ctx = Vbf + plane;

    const dim3 blk(128);
    const dim3 ggrid(D_MODEL / 64, M / 64);

    gemm512_kernel<false, false><<<ggrid, blk, 0, stream>>>(queries, nullptr, Wq, bq, Qbf, nullptr);
    gemm512_kernel<false, false><<<ggrid, blk, 0, stream>>>(keys,    nullptr, Wk, bk, Kbf, nullptr);
    gemm512_kernel<false, false><<<ggrid, blk, 0, stream>>>(values,  nullptr, Wv, bv, Vbf, nullptr);

    const dim3 agrid(SEQ / 64, NHEADS, BATCH);
    attn_kernel<<<agrid, blk, 0, stream>>>(Qbf, Kbf, Vbf, tau, delta, Ctx);

    gemm512_kernel<true, true><<<ggrid, blk, 0, stream>>>(nullptr, Ctx, Wo, bo, nullptr, out);
}
